// edge_block_80857054314755
// MI455X (gfx1250) — matrix operand order repaired and checked by reading, not compiled, not run on hardware
//
#include <hip/hip_runtime.h>
#include <cstdint>

constexpr int DIMC = 96;
constexpr int CI   = 192;
constexpr int BB   = 2;
constexpr int HH   = 128;
constexpr int WW   = 128;
constexpr int LHW  = HH * WW;     // 16384
constexpr int NPIX = BB * LHW;    // 32768

typedef __attribute__((ext_vector_type(16))) _Float16 v16h;
typedef __attribute__((ext_vector_type(8)))  _Float16 v8h;
typedef __attribute__((ext_vector_type(8)))  float    v8f;

// direction remap: scan position l -> spatial index in [0, LHW)
__device__ __forceinline__ int dir_map(int dir, int l) {
    if (dir <= 0) return l;
    if (dir == 2) return LHW - 1 - l;
    int lr = (dir == 3) ? (LHW - 1 - l) : l;
    int w = lr >> 7;        // lr / HH
    int h = lr & (HH - 1);  // lr % HH
    return h * WW + w;
}

// ---------------------------------------------------------------------------
// Weight prep: fp32 (M,K) row-major -> f16, zero-padded to (Mpad, Kpad)
// ---------------------------------------------------------------------------
__global__ __launch_bounds__(256) void prep_w_kernel(
    const float* __restrict__ w, _Float16* __restrict__ out,
    int M, int K, int Kpad, int Mpad)
{
    int i = blockIdx.x * blockDim.x + threadIdx.x;
    int n = Mpad * Kpad;
    if (i >= n) return;
    int o = i / Kpad, k = i % Kpad;
    float v = (o < M && k < K) ? w[(size_t)o * K + k] : 0.0f;
    out[i] = (_Float16)v;
}

// ---------------------------------------------------------------------------
// BN fold: scale = g*rsqrt(v+eps); shift = b + (bias - m)*scale
// ---------------------------------------------------------------------------
__global__ __launch_bounds__(128) void bn_prep_kernel(
    const float* __restrict__ g, const float* __restrict__ b,
    const float* __restrict__ m, const float* __restrict__ v,
    const float* __restrict__ bias,
    float* __restrict__ scale, float* __restrict__ shift, int C)
{
    int i = threadIdx.x;
    if (i >= C) return;
    float s = g[i] * rsqrtf(v[i] + 1e-5f);
    scale[i] = s;
    shift[i] = b[i] + ((bias ? bias[i] : 0.0f) - m[i]) * s;
}

// ---------------------------------------------------------------------------
// LayerNorm over channel dim, per pixel. x,y: [B, C, L]
// ---------------------------------------------------------------------------
__global__ __launch_bounds__(256) void ln2d_kernel(
    const float* __restrict__ x, const float* __restrict__ g,
    const float* __restrict__ bp, float* __restrict__ y, int C)
{
    int p = blockIdx.x * blockDim.x + threadIdx.x;
    if (p >= NPIX) return;
    int b = p >> 14, l = p & (LHW - 1);
    const float* base = x + (size_t)b * C * LHW + l;
    float s = 0.0f, s2 = 0.0f;
    for (int c = 0; c < C; ++c) {
        float v = base[(size_t)c * LHW];
        s += v; s2 += v * v;
    }
    float mu = s / C;
    float var = s2 / C - mu * mu;
    float r = rsqrtf(var + 1e-5f);
    float* yb = y + (size_t)b * C * LHW + l;
    for (int c = 0; c < C; ++c) {
        float v = base[(size_t)c * LHW];
        yb[(size_t)c * LHW] = (v - mu) * r * g[c] + bp[c];
    }
}

// ---------------------------------------------------------------------------
// Fused WMMA GEMM:  Y[o,p] = epi( sum_k W[o,k] * X(k,p) )
//
// Branchless loader: weights are zero-padded to (Mpad,Kpad), so padded K
// channels read readable scratch multiplied by zero weights; all Mpad rows
// are stored (padded rows accumulate exactly zero).  Split-K concat (aff
// block) is one pointer select per K group (Ksplit is 16-aligned).
//
// Each wave computes a 16x64 output tile: one A fragment (weights) reused
// against four B fragments -> 4 x v_wmma_f32_16x16x32_f16 per K step.
// ---------------------------------------------------------------------------
constexpr int NT = 4;   // N sub-tiles per wave

__global__ __launch_bounds__(256) void wmma_gemm_kernel(
    const _Float16* __restrict__ Wh, int Kpad, int Ksplit,
    const float* __restrict__ X0, int C0,
    const float* __restrict__ X1, int C1,
    int dir,
    float* __restrict__ Y, int Mtot,
    const float* __restrict__ scale, const float* __restrict__ shift,
    const float* __restrict__ bias, int act,
    const float* __restrict__ res, int Mtiles)
{
    int gwave = (blockIdx.x * blockDim.x + threadIdx.x) >> 5;
    int lane  = threadIdx.x & 31;
    int total = Mtiles * (NPIX / (16 * NT));
    if (gwave >= total) return;
    int mtile = gwave % Mtiles;
    int ntile = gwave / Mtiles;
    int p0 = ntile * 16 * NT;
    int bidx = p0 >> 14;            // batch (L multiple of 64 -> no straddle)
    int l0 = p0 & (LHW - 1);
    int hl = lane >> 4;             // lane half
    int mrow = mtile * 16 + (lane & 15);
    int ncol = lane & 15;

    const _Float16* wrow = Wh + (size_t)mrow * Kpad;
    const float* x0t[NT];
    const float* x1t[NT];
#pragma unroll
    for (int t = 0; t < NT; ++t) {
        int ls = dir_map(dir, l0 + 16 * t + ncol);
        x0t[t] = X0 + (size_t)bidx * C0 * LHW + ls;
        x1t[t] = X1 ? (X1 + (size_t)bidx * C1 * LHW + ls) : x0t[t];
    }

    v8f acc[NT];
#pragma unroll
    for (int t = 0; t < NT; ++t)
#pragma unroll
        for (int i = 0; i < 8; ++i) acc[t][i] = 0.0f;

    for (int k0 = 0; k0 < Kpad; k0 += 32) {
        // A fragment (16x32 f16): lane m = mrow; elems 0..15 at K=k0+16*hl+e
        // -> two contiguous 16B loads.
        v8h lo = *(const v8h*)(wrow + k0 + 16 * hl);
        v8h hi = *(const v8h*)(wrow + k0 + 16 * hl + 8);
        v16h afrag;
#pragma unroll
        for (int e = 0; e < 8; ++e) { afrag[e] = lo[e]; afrag[8 + e] = hi[e]; }
        __builtin_prefetch(wrow + k0 + 32, 0, 3);

        // B fragments (32x16 f16): n = lane%16, K = k0 + 16*hl + e.
        // One pointer select per K group, then loads at constant offsets
        // (coalesced across lanes for each e).
        int kg = k0 + 16 * hl;
        bool s0 = kg < Ksplit;
        size_t off0 = (size_t)kg * LHW;
        size_t off1 = (size_t)(kg - Ksplit) * LHW;
#pragma unroll
        for (int t = 0; t < NT; ++t) {
            const float* src = s0 ? (x0t[t] + off0) : (x1t[t] + off1);
            v16h bfrag;
#pragma unroll
            for (int e = 0; e < 16; ++e)
                bfrag[e] = (_Float16)src[(size_t)e * LHW];
            acc[t] = __builtin_amdgcn_wmma_f32_16x16x32_f16(
                false, afrag, false, bfrag, (short)0, acc[t], false, false);
        }
    }

    // C/D layout: elem v -> (m = mtile*16 + v + 8*hl, n = lane%16).
    // All Mpad rows stored unconditionally (padded rows have acc == 0).
#pragma unroll
    for (int v = 0; v < 8; ++v) {
        int o = mtile * 16 + v + 8 * hl;
        float sc = scale ? scale[o] : 1.0f;
        float sh = scale ? shift[o] : (bias ? bias[o] : 0.0f);
#pragma unroll
        for (int t = 0; t < NT; ++t) {
            float tv = acc[t][v] * sc + sh;
            if (act == 1) {
                tv = fmaxf(tv, 0.0f);
            } else if (act == 2) {         // gelu (tanh approx, jax default)
                float x3 = tv * tv * tv;
                tv = 0.5f * tv * (1.0f + tanhf(0.7978845608028654f * (tv + 0.044715f * x3)));
            } else if (act == 3) {         // silu
                tv = tv / (1.0f + __expf(-tv));
            }
            size_t oi = ((size_t)bidx * Mtot + o) * LHW
                      + (size_t)(l0 + 16 * t + ncol);
            if (res) tv += res[oi];
            Y[oi] = tv;
        }
    }
}

// ---------------------------------------------------------------------------
// 3x3 depthwise conv (SAME, zero pad) + SiLU.  x,y: [B, CI, L]
// ---------------------------------------------------------------------------
__global__ __launch_bounds__(256) void dwconv3_silu_kernel(
    const float* __restrict__ x, const float* __restrict__ wt, float* __restrict__ y)
{
    size_t i = (size_t)blockIdx.x * blockDim.x + threadIdx.x;
    if (i >= (size_t)CI * NPIX) return;
    int l = (int)(i & (LHW - 1));
    int bc = (int)(i >> 14);
    int c = bc % CI;
    int h = l >> 7, w = l & (WW - 1);
    const float* xb = x + (size_t)bc * LHW;
    const float* wk = wt + c * 9;
    float acc = 0.0f;
#pragma unroll
    for (int dy = -1; dy <= 1; ++dy)
#pragma unroll
        for (int dx = -1; dx <= 1; ++dx) {
            int hh = h + dy, ww2 = w + dx;
            if (hh >= 0 && hh < HH && ww2 >= 0 && ww2 < WW)
                acc += wk[(dy + 1) * 3 + (dx + 1)] * xb[hh * WW + ww2];
        }
    y[i] = acc / (1.0f + __expf(-acc));
}

// ---------------------------------------------------------------------------
// Selective scan: one wave per (dir, b, c) row; 32-wide Hillis-Steele scan of
// (a,b) pair composition with scalar carry; fused dt projection/softplus/exp;
// atomic scatter-accumulate of the 4 directions into ym.
// proj rows have stride 16*LHW per (dir,b) (padded GEMM output).
// ---------------------------------------------------------------------------
__global__ __launch_bounds__(256) void ss2d_scan_kernel(
    const float* __restrict__ u, const float* __restrict__ proj,
    const float* __restrict__ dt_w, const float* __restrict__ dt_b,
    const float* __restrict__ A_log, const float* __restrict__ Dp,
    float* __restrict__ ym)
{
    int gwave = (blockIdx.x * blockDim.x + threadIdx.x) >> 5;
    int lane = threadIdx.x & 31;
    if (gwave >= 4 * BB * CI) return;
    int c = gwave % CI;
    int t = gwave / CI;
    int b = t % BB;
    int dir = t / BB;

    float wv[6];
#pragma unroll
    for (int r = 0; r < 6; ++r) wv[r] = dt_w[((size_t)dir * CI + c) * 6 + r];
    float dtb = dt_b[dir * CI + c];
    float Aa = -__expf(A_log[dir * CI + c]);
    float Dv = Dp[dir * CI + c];

    const float* pb = proj + (size_t)(dir * BB + b) * 16 * LHW;
    const float* ub = u + ((size_t)b * CI + c) * LHW;
    float* yb = ym + ((size_t)b * CI + c) * LHW;

    float carry = 0.0f;
    for (int blk = 0; blk < LHW; blk += 32) {
        int l = blk + lane;
        int lsrc = dir_map(dir, l);
        float x = ub[lsrc];
        float z = dtb;
#pragma unroll
        for (int r = 0; r < 6; ++r) z += wv[r] * pb[(size_t)r * LHW + l];
        float dt = (z > 20.0f) ? z : log1pf(__expf(z));   // softplus
        float Bs = pb[(size_t)6 * LHW + l];
        float Cs = pb[(size_t)7 * LHW + l];
        float a = __expf(dt * Aa);
        float bbv = dt * Bs * x;
        // inclusive scan of pair composition (a1,b1)o(a2,b2)=(a1*a2, b1*a2+b2)
#pragma unroll
        for (int d = 1; d < 32; d <<= 1) {
            float pa = __shfl_up(a, d, 32);
            float pbv = __shfl_up(bbv, d, 32);
            if (lane >= d) { bbv = pbv * a + bbv; a = pa * a; }
        }
        float h = a * carry + bbv;
        float y = h * Cs + x * Dv;
        atomicAdd(&yb[lsrc], y);
        carry = __shfl(h, 31, 32);
    }
}

// ---------------------------------------------------------------------------
// Difference-of-Gaussians: 5x5 stencil with (k1 - k2), SAME zero pad.
// g: [B, 16, L] (padded), channel 0 valid.
// ---------------------------------------------------------------------------
__global__ __launch_bounds__(256) void dog_kernel(
    const float* __restrict__ g, const float* __restrict__ k1,
    const float* __restrict__ k2, float* __restrict__ out)
{
    int i = blockIdx.x * blockDim.x + threadIdx.x;
    if (i >= NPIX) return;
    int l = i & (LHW - 1);
    int b = i >> 14;
    int h = l >> 7, w = l & (WW - 1);
    const float* gb = g + (size_t)b * 16 * LHW;
    float acc = 0.0f;
    for (int ky = 0; ky < 5; ++ky)
        for (int kx = 0; kx < 5; ++kx) {
            int hh = h + ky - 2, ww2 = w + kx - 2;
            if (hh >= 0 && hh < HH && ww2 >= 0 && ww2 < WW)
                acc += (k1[ky * 5 + kx] - k2[ky * 5 + kx]) * gb[hh * WW + ww2];
        }
    out[i] = acc;
}

// ---------------------------------------------------------------------------
// Build 288-channel concat [Di*xv, dog*xv, Lap*xv]
// ---------------------------------------------------------------------------
__global__ __launch_bounds__(256) void cat288_kernel(
    const float* __restrict__ Di, const float* __restrict__ Lap,
    const float* __restrict__ dog, const float* __restrict__ xv,
    float* __restrict__ out)
{
    size_t i = (size_t)blockIdx.x * blockDim.x + threadIdx.x;
    if (i >= (size_t)3 * DIMC * NPIX) return;
    int l = (int)(i & (LHW - 1));
    int bc = (int)(i >> 14);
    int c = bc % (3 * DIMC);
    int b = bc / (3 * DIMC);
    float m = xv[((size_t)b * DIMC + (c % DIMC)) * LHW + l];
    float s;
    if (c < DIMC)          s = Di[((size_t)b * DIMC + c) * LHW + l];
    else if (c < 2 * DIMC) s = dog[(size_t)b * LHW + l];
    else                   s = Lap[((size_t)b * DIMC + (c - 2 * DIMC)) * LHW + l];
    out[i] = s * m;
}

// ---------------------------------------------------------------------------
// Final affine blend: d = 1 + tanh(dbuf); out = x1*d + x2*(2-d)
// ---------------------------------------------------------------------------
__global__ __launch_bounds__(256) void aff_final_kernel(
    const float* __restrict__ x1, const float* __restrict__ x2,
    const float* __restrict__ dbuf, float* __restrict__ out)
{
    size_t i = (size_t)blockIdx.x * blockDim.x + threadIdx.x;
    if (i >= (size_t)DIMC * NPIX) return;
    float d = 1.0f + tanhf(dbuf[i]);
    out[i] = x1[i] * d + x2[i] * (2.0f - d);
}

// ---------------------------------------------------------------------------
// Host orchestration
// ---------------------------------------------------------------------------
extern "C" void kernel_launch(void* const* d_in, const int* in_sizes, int n_in,
                              void* d_out, int out_size, void* d_ws, size_t ws_size,
                              hipStream_t stream)
{
    (void)in_sizes; (void)n_in; (void)out_size; (void)ws_size;
    auto F = [&](int i) -> const float* { return (const float*)d_in[i]; };

    const float* AFFout  = F(0);
    const float* ABfuse  = F(1);
    const float* Di      = F(2);
    const float* Laplace = F(3);
    // param bases: vssb1=4, vssb2=22, gray=40, fuse=50, aff=60, dog=72/73

    // ---- workspace bump allocator -----------------------------------------
    uintptr_t cur = (uintptr_t)d_ws;
    auto alloc = [&](size_t bytes) -> void* {
        cur = (cur + 255) & ~(uintptr_t)255;
        void* p = (void*)cur; cur += bytes; return p;
    };
    float* bufA   = (float*)alloc((size_t)DIMC * NPIX * 4);  // also 32/80-ch padded use
    float* bufB   = (float*)alloc((size_t)DIMC * NPIX * 4);
    float* bufC   = (float*)alloc((size_t)DIMC * NPIX * 4);
    float* xv1    = (float*)alloc((size_t)DIMC * NPIX * 4);
    float* t192a  = (float*)alloc((size_t)CI * NPIX * 4);
    float* t192b  = (float*)alloc((size_t)CI * NPIX * 4);
    float* t384   = (float*)alloc((size_t)4 * DIMC * NPIX * 4);
    float* projB  = (float*)alloc((size_t)4 * BB * 16 * LHW * 4);  // padded M=16
    float* g1buf  = (float*)alloc((size_t)BB * 16 * LHW * 4);      // padded M=16
    float* dogbuf = (float*)alloc((size_t)NPIX * 4);
    float* fpool  = (float*)alloc(4096);
    _Float16* hpool = (_Float16*)alloc(600000);   // 300000 halfs (+ prefetch slack)

    size_t ho = 0;
    auto halloc = [&](int M, int K) -> _Float16* {
        int Mp = (M + 15) / 16 * 16, Kp = (K + 31) / 32 * 32;
        _Float16* p = hpool + ho; ho += (size_t)Mp * Kp; return p;
    };
    auto prepw = [&](const float* w, _Float16* dst, int M, int K) {
        int Mp = (M + 15) / 16 * 16, Kp = (K + 31) / 32 * 32;
        int n = Mp * Kp;
        prep_w_kernel<<<(n + 255) / 256, 256, 0, stream>>>(w, dst, M, K, Kp, Mp);
    };
    // M = logical out channels, Mtot = padded channel stride of Y
    auto gemm = [&](const _Float16* Wh, int M, int K,
                    const float* X0, int C0, const float* X1, int C1, int ksplit,
                    int dir, float* Y, int Mtot,
                    const float* scale, const float* shift, const float* bias,
                    int act, const float* res) {
        int Mtiles = (M + 15) / 16;
        int Kp = (K + 31) / 32 * 32;
        int ks = (ksplit < Kp) ? ksplit : Kp;
        int waves = Mtiles * (NPIX / (16 * NT));
        wmma_gemm_kernel<<<(waves + 7) / 8, 256, 0, stream>>>(
            Wh, Kp, ks, X0, C0, X1, C1, dir, Y, Mtot,
            scale, shift, bias, act, res, Mtiles);
    };

    // ---- f16 weight prep ---------------------------------------------------
    _Float16* b1_in  = halloc(CI, DIMC);
    _Float16* b1_xp  = hpool + ho; ho += (size_t)4 * 16 * CI;
    _Float16* b1_out = halloc(DIMC, CI);
    _Float16* b1_m1  = halloc(4 * DIMC, DIMC);
    _Float16* b1_m2  = halloc(DIMC, 4 * DIMC);
    _Float16* b2_in  = halloc(CI, DIMC);
    _Float16* b2_xp  = hpool + ho; ho += (size_t)4 * 16 * CI;
    _Float16* b2_out = halloc(DIMC, CI);
    _Float16* b2_m1  = halloc(4 * DIMC, DIMC);
    _Float16* b2_m2  = halloc(DIMC, 4 * DIMC);
    _Float16* g_w1   = halloc(24, DIMC);
    _Float16* g_w2   = halloc(1, 24);
    _Float16* f_w1   = halloc(72, 3 * DIMC);
    _Float16* f_w2   = halloc(DIMC, 72);
    _Float16* a_w1   = halloc(24, 2 * DIMC);
    _Float16* a_w2   = halloc(DIMC, 24);

    prepw(F(4 + 2), b1_in, CI, DIMC);
    for (int d = 0; d < 4; ++d)
        prepw(F(4 + 4) + (size_t)d * 8 * CI, b1_xp + (size_t)d * 16 * CI, 8, CI);
    prepw(F(4 + 11), b1_out, DIMC, CI);
    prepw(F(4 + 14), b1_m1, 4 * DIMC, DIMC);
    prepw(F(4 + 16), b1_m2, DIMC, 4 * DIMC);
    prepw(F(22 + 2), b2_in, CI, DIMC);
    for (int d = 0; d < 4; ++d)
        prepw(F(22 + 4) + (size_t)d * 8 * CI, b2_xp + (size_t)d * 16 * CI, 8, CI);
    prepw(F(22 + 11), b2_out, DIMC, CI);
    prepw(F(22 + 14), b2_m1, 4 * DIMC, DIMC);
    prepw(F(22 + 16), b2_m2, DIMC, 4 * DIMC);
    prepw(F(40), g_w1, 24, DIMC);
    prepw(F(45), g_w2, 1, 24);
    prepw(F(50), f_w1, 72, 3 * DIMC);
    prepw(F(55), f_w2, DIMC, 72);
    prepw(F(60), a_w1, 24, 2 * DIMC);
    prepw(F(66), a_w2, DIMC, 24);

    // ---- BN scale/shift folds (padded reads land inside fpool) ------------
    float *g1s = fpool + 0,   *g1h = fpool + 32;
    float *g2s = fpool + 64,  *g2h = fpool + 96;
    float *f1s = fpool + 128, *f1h = fpool + 256;
    float *f2s = fpool + 384, *f2h = fpool + 480;
    float *a1s = fpool + 576, *a1h = fpool + 608;
    float *a2s = fpool + 640, *a2h = fpool + 736;
    bn_prep_kernel<<<1, 128, 0, stream>>>(F(41), F(42), F(43), F(44), nullptr, g1s, g1h, 24);
    bn_prep_kernel<<<1, 128, 0, stream>>>(F(46), F(47), F(48), F(49), nullptr, g2s, g2h, 1);
    bn_prep_kernel<<<1, 128, 0, stream>>>(F(51), F(52), F(53), F(54), nullptr, f1s, f1h, 72);
    bn_prep_kernel<<<1, 128, 0, stream>>>(F(56), F(57), F(58), F(59), nullptr, f2s, f2h, DIMC);
    bn_prep_kernel<<<1, 128, 0, stream>>>(F(62), F(63), F(64), F(65), F(61), a1s, a1h, 24);
    bn_prep_kernel<<<1, 128, 0, stream>>>(F(68), F(69), F(70), F(71), F(67), a2s, a2h, DIMC);

    const int NB = NPIX / 256;

    // ---- VSS block ---------------------------------------------------------
    auto run_vssb = [&](int pb, const float* xin, float* xout,
                        const _Float16* inph, const _Float16* xph,
                        const _Float16* outh, const _Float16* m1h,
                        const _Float16* m2h) {
        // ln1 -> in_proj GEMM -> dwconv3+silu
        ln2d_kernel<<<NB, 256, 0, stream>>>(xin, F(pb + 0), F(pb + 1), bufB, DIMC);
        gemm(inph, CI, DIMC, bufB, DIMC, nullptr, 0, 1 << 30, -1, t192a, CI,
             nullptr, nullptr, nullptr, 0, nullptr);
        dwconv3_silu_kernel<<<(int)(((size_t)CI * NPIX + 255) / 256), 256, 0, stream>>>(
            t192a, F(pb + 3), t192b);
        // x_proj: 4 directions, gathered columns of u; padded M=16 rows
        for (int d = 0; d < 4; ++d)
            gemm(xph + (size_t)d * 16 * CI, 8, CI, t192b, CI, nullptr, 0, 1 << 30, d,
                 projB + (size_t)d * BB * 16 * LHW, 16,
                 nullptr, nullptr, nullptr, 0, nullptr);
        // selective scan, atomic scatter into ym (t192a)
        hipMemsetAsync(t192a, 0, (size_t)CI * NPIX * sizeof(float), stream);
        ss2d_scan_kernel<<<192, 256, 0, stream>>>(t192b, projB, F(pb + 5),
                                                  F(pb + 6), F(pb + 7), F(pb + 8), t192a);
        // out-norm -> out_proj (+ residual xin)
        ln2d_kernel<<<NB, 256, 0, stream>>>(t192a, F(pb + 9), F(pb + 10), t192b, CI);
        gemm(outh, DIMC, CI, t192b, CI, nullptr, 0, 1 << 30, -1, bufA, DIMC,
             nullptr, nullptr, nullptr, 0, xin);
        // MLP: ln2 -> gelu GEMM -> GEMM (+ residual bufA)
        ln2d_kernel<<<NB, 256, 0, stream>>>(bufA, F(pb + 12), F(pb + 13), bufB, DIMC);
        gemm(m1h, 4 * DIMC, DIMC, bufB, DIMC, nullptr, 0, 1 << 30, -1, t384, 4 * DIMC,
             nullptr, nullptr, F(pb + 15), 2, nullptr);
        gemm(m2h, DIMC, 4 * DIMC, t384, 4 * DIMC, nullptr, 0, 1 << 30, -1, xout, DIMC,
             nullptr, nullptr, F(pb + 17), 0, bufA);
    };

    // Stage 1: AB_VSS = vssblock1(AFFout)
    run_vssb(4, AFFout, xv1, b1_in, b1_xp, b1_out, b1_m1, b1_m2);

    // Stage 2: gray path + DoG + concat + fuse  (padded channel strides)
    gemm(g_w1, 24, DIMC, ABfuse, DIMC, nullptr, 0, 1 << 30, -1, bufA, 32,
         g1s, g1h, nullptr, 1, nullptr);
    gemm(g_w2, 1, 24, bufA, 32, nullptr, 0, 1 << 30, -1, g1buf, 16,
         g2s, g2h, nullptr, 0, nullptr);
    dog_kernel<<<NB, 256, 0, stream>>>(g1buf, F(72), F(73), dogbuf);
    cat288_kernel<<<(int)(((size_t)3 * DIMC * NPIX + 255) / 256), 256, 0, stream>>>(
        Di, Laplace, dogbuf, xv1, t384);
    gemm(f_w1, 72, 3 * DIMC, t384, 3 * DIMC, nullptr, 0, 1 << 30, -1, bufA, 80,
         f1s, f1h, nullptr, 1, nullptr);
    gemm(f_w2, DIMC, 72, bufA, 80, nullptr, 0, 1 << 30, -1, bufC, DIMC,
         f2s, f2h, nullptr, 0, nullptr);

    // Stage 3: c = vssblock2(bufC) -> bufB
    run_vssb(22, bufC, bufB, b2_in, b2_xp, b2_out, b2_m1, b2_m2);

    // Stage 4: affine fusion of (xv1, bufB); split-K concat in the GEMM loader
    gemm(a_w1, 24, 2 * DIMC, xv1, DIMC, bufB, DIMC, DIMC, -1, bufA, 32,
         a1s, a1h, nullptr, 3, nullptr);
    gemm(a_w2, DIMC, 24, bufA, 32, nullptr, 0, 1 << 30, -1, t192a, DIMC,
         a2s, a2h, nullptr, 0, nullptr);
    aff_final_kernel<<<(int)(((size_t)DIMC * NPIX + 255) / 256), 256, 0, stream>>>(
        xv1, bufB, t192a, (float*)d_out);
}
